// Reasoner_25056839205792
// MI455X (gfx1250) — compile-verified, matrix operand order checked by reading, not run on hardware
//
#include <hip/hip_runtime.h>
#include <hip/hip_bf16.h>
#include <math.h>

#define T_SEQ 1024
#define D_MODEL 768
#define NLAYER 16
#define NHEAD 12
#define HDIM 64
#define FFDIM 3072
#define RMS_EPS 1e-6f

typedef __attribute__((ext_vector_type(8)))  __bf16 v8bf;
typedef __attribute__((ext_vector_type(16))) __bf16 v16bf;
typedef __attribute__((ext_vector_type(8)))  float  v8f;

// float -> bf16 round-to-nearest-even via bits
static __device__ __forceinline__ __bf16 f2bf(float f) {
    unsigned u = __float_as_uint(f);
    unsigned r = (u + 0x7FFFu + ((u >> 16) & 1u)) >> 16;
    union { unsigned short s; __bf16 b; } cv;
    cv.s = (unsigned short)r;
    return cv.b;
}

// assemble a 16-element bf16 fragment from two 16-byte chunks
static __device__ __forceinline__ v16bf frag16(const __bf16* p0, const __bf16* p1) {
    union { v16bf w; v8bf h[2]; } u;
    u.h[0] = *(const v8bf*)p0;
    u.h[1] = *(const v8bf*)p1;
    return u.w;
}

// low 32 bits of a generic pointer into __shared__ are the wave-relative LDS
// byte offset (ISA: LDS aperture addresses map to LDS by discarding upper bits)
static __device__ __forceinline__ unsigned lds_off(const void* p) {
    return (unsigned)(size_t)p;
}

// async 16B copy global -> LDS (CDNA5 Tensor/async path, tracked by ASYNCcnt)
static __device__ __forceinline__ void async_cp16(unsigned lds_addr, const void* gaddr) {
    asm volatile("global_load_async_to_lds_b128 %0, %1, off"
                 :: "v"(lds_addr), "v"(gaddr) : "memory");
}
static __device__ __forceinline__ void wait_async0() {
    asm volatile("s_wait_asynccnt 0" ::: "memory");
}

// ---------------------------------------------------------------------------
// fp32 -> bf16 converter (8 elems / thread, 16B stores)
// ---------------------------------------------------------------------------
__global__ void cvt_bf16(const float* __restrict__ in, __bf16* __restrict__ out, long n8) {
    long i = (long)blockIdx.x * blockDim.x + threadIdx.x;
    if (i >= n8) return;
    const float4* p = (const float4*)in + i * 2;
    float4 a = p[0], b = p[1];
    union { __bf16 h[8]; uint4 u; } o;
    o.h[0] = f2bf(a.x); o.h[1] = f2bf(a.y); o.h[2] = f2bf(a.z); o.h[3] = f2bf(a.w);
    o.h[4] = f2bf(b.x); o.h[5] = f2bf(b.y); o.h[6] = f2bf(b.z); o.h[7] = f2bf(b.w);
    ((uint4*)out)[i] = o.u;
}

// ---------------------------------------------------------------------------
// x = emb[token_ids] + pos_emb
// ---------------------------------------------------------------------------
__global__ void embed_k(const int* __restrict__ ids, const float* __restrict__ emb,
                        const float* __restrict__ pos, float* __restrict__ X) {
    int gid = blockIdx.x * 256 + threadIdx.x;
    int t = gid / D_MODEL, d = gid - t * D_MODEL;
    X[gid] = emb[(size_t)ids[t] * D_MODEL + d] + pos[gid];
}

// ---------------------------------------------------------------------------
// RMSNorm (per token) -> bf16
// ---------------------------------------------------------------------------
__global__ __launch_bounds__(256)
void rmsnorm_bf16(const float* __restrict__ X, const float* __restrict__ W,
                  __bf16* __restrict__ Hout) {
    __shared__ float red[8];
    int t = blockIdx.x, tid = threadIdx.x;
    const float* xr = X + (size_t)t * D_MODEL;
    float v0 = xr[tid], v1 = xr[tid + 256], v2 = xr[tid + 512];
    float ss = v0 * v0 + v1 * v1 + v2 * v2;
    #pragma unroll
    for (int o = 16; o; o >>= 1) ss += __shfl_xor(ss, o);
    if ((tid & 31) == 0) red[tid >> 5] = ss;
    __syncthreads();
    float tot = 0.f;
    #pragma unroll
    for (int i = 0; i < 8; ++i) tot += red[i];
    float inv = rsqrtf(tot * (1.f / (float)D_MODEL) + RMS_EPS);
    __bf16* hr = Hout + (size_t)t * D_MODEL;
    hr[tid]       = f2bf(v0 * inv * W[tid]);
    hr[tid + 256] = f2bf(v1 * inv * W[tid + 256]);
    hr[tid + 512] = f2bf(v2 * inv * W[tid + 512]);
}

// per-token inv-RMS only (for final norm)
__global__ __launch_bounds__(256)
void invrms_k(const float* __restrict__ X, float* __restrict__ inv) {
    __shared__ float red[8];
    int t = blockIdx.x, tid = threadIdx.x;
    const float* xr = X + (size_t)t * D_MODEL;
    float v0 = xr[tid], v1 = xr[tid + 256], v2 = xr[tid + 512];
    float ss = v0 * v0 + v1 * v1 + v2 * v2;
    #pragma unroll
    for (int o = 16; o; o >>= 1) ss += __shfl_xor(ss, o);
    if ((tid & 31) == 0) red[tid >> 5] = ss;
    __syncthreads();
    float tot = 0.f;
    #pragma unroll
    for (int i = 0; i < 8; ++i) tot += red[i];
    if (tid == 0) inv[t] = rsqrtf(tot * (1.f / (float)D_MODEL) + RMS_EPS);
}

// ---------------------------------------------------------------------------
// WMMA GEMM:  C(M x N) = A(M x K, bf16, row-major) @ B(N x K, bf16, row-major)^T
// tile 128x64, BK=32, 8 waves each 32x32 (2x2 WMMA frags).
// Double-buffered LDS filled with global_load_async_to_lds_b128 (ASYNCcnt).
// EPI: 0 = store f32, 1 = f32 residual add, 2 = exact GELU -> bf16
// ---------------------------------------------------------------------------
#define LDST 40
template<int EPI>
__global__ __launch_bounds__(256)
void gemm_bf16(const __bf16* __restrict__ A, const __bf16* __restrict__ B,
               float* __restrict__ Cf, __bf16* __restrict__ Cb,
               const float* __restrict__ Res, int N, int K) {
    __shared__ __bf16 sA[2][128 * LDST];
    __shared__ __bf16 sB[2][64 * LDST];
    const int tid = threadIdx.x;
    const int lane = tid & 31, wave = tid >> 5;
    const int wm = wave & 3, wn = wave >> 2;          // 4 x 2 wave grid
    const int hl = lane >> 4, l16 = lane & 15;
    const int bM = blockIdx.y * 128, bN = blockIdx.x * 64;
    const int lr = tid >> 2;                          // 0..63
    const int lc = (tid & 3) * 8;                     // 0,8,16,24

    v8f acc[2][2];
    #pragma unroll
    for (int mi = 0; mi < 2; ++mi)
        #pragma unroll
        for (int ni = 0; ni < 2; ++ni)
            #pragma unroll
            for (int j = 0; j < 8; ++j) acc[mi][ni][j] = 0.f;

    auto issue = [&](int buf, int k0) {
        async_cp16(lds_off(&sA[buf][lr * LDST + lc]),        &A[(size_t)(bM + lr) * K + k0 + lc]);
        async_cp16(lds_off(&sA[buf][(lr + 64) * LDST + lc]), &A[(size_t)(bM + lr + 64) * K + k0 + lc]);
        async_cp16(lds_off(&sB[buf][lr * LDST + lc]),        &B[(size_t)(bN + lr) * K + k0 + lc]);
    };

    const int nk = K >> 5;
    issue(0, 0);
    for (int i = 0; i < nk; ++i) {
        const int buf = i & 1;
        wait_async0();                 // this wave's async fills complete
        __syncthreads();               // all waves' fills complete -> LDS valid
        if (i + 1 < nk) issue(buf ^ 1, (i + 1) * 32);

        v16bf af[2], bfr[2];
        #pragma unroll
        for (int mi = 0; mi < 2; ++mi) {
            int r = wm * 32 + mi * 16 + l16;
            af[mi] = frag16(&sA[buf][r * LDST + hl * 8], &sA[buf][r * LDST + 16 + hl * 8]);
        }
        #pragma unroll
        for (int ni = 0; ni < 2; ++ni) {
            int r = wn * 32 + ni * 16 + l16;
            bfr[ni] = frag16(&sB[buf][r * LDST + hl * 8], &sB[buf][r * LDST + 16 + hl * 8]);
        }
        #pragma unroll
        for (int mi = 0; mi < 2; ++mi)
            #pragma unroll
            for (int ni = 0; ni < 2; ++ni)
                acc[mi][ni] = __builtin_amdgcn_wmma_f32_16x16x32_bf16(
                    false, af[mi], false, bfr[ni], (short)0, acc[mi][ni], false, false);
        __syncthreads();               // reads of buf done before it is refilled
    }

    #pragma unroll
    for (int mi = 0; mi < 2; ++mi)
        #pragma unroll
        for (int ni = 0; ni < 2; ++ni) {
            int mBase = bM + wm * 32 + mi * 16 + (hl ? 8 : 0);
            int n = bN + wn * 32 + ni * 16 + l16;
            #pragma unroll
            for (int j = 0; j < 8; ++j) {
                size_t idx = (size_t)(mBase + j) * N + n;
                float v = acc[mi][ni][j];
                if (EPI == 0) {
                    Cf[idx] = v;
                } else if (EPI == 1) {
                    Cf[idx] = Res[idx] + v;
                } else {
                    float g = 0.5f * v * (1.f + erff(v * 0.70710678118654752f));
                    Cb[idx] = f2bf(g);
                }
            }
        }
}

// ---------------------------------------------------------------------------
// QKV post: per-(head,token) L2-normalize q,k -> bf16 (H,T,64); v -> bf16 (H,64,T)
// ---------------------------------------------------------------------------
__global__ __launch_bounds__(64)
void qkv_post(const float* __restrict__ QKV, __bf16* __restrict__ Qh,
              __bf16* __restrict__ Kh, __bf16* __restrict__ Vt) {
    __shared__ float rq[2], rk[2];
    int b = blockIdx.x;
    int t = b / NHEAD, h = b % NHEAD;
    int d = threadIdx.x;
    const float* row = QKV + (size_t)t * (3 * D_MODEL);
    float qv = row[h * HDIM + d];
    float kv = row[D_MODEL + h * HDIM + d];
    float vv = row[2 * D_MODEL + h * HDIM + d];
    float sq = qv * qv, sk = kv * kv;
    #pragma unroll
    for (int o = 16; o; o >>= 1) { sq += __shfl_xor(sq, o); sk += __shfl_xor(sk, o); }
    if ((d & 31) == 0) { rq[d >> 5] = sq; rk[d >> 5] = sk; }
    __syncthreads();
    float nq = fmaxf(sqrtf(rq[0] + rq[1]), 1e-12f);
    float nk = fmaxf(sqrtf(rk[0] + rk[1]), 1e-12f);
    size_t qi = ((size_t)h * T_SEQ + t) * HDIM + d;
    Qh[qi] = f2bf(qv / nq);
    Kh[qi] = f2bf(kv / nk);
    Vt[((size_t)h * HDIM + d) * T_SEQ + t] = f2bf(vv);
}

// ---------------------------------------------------------------------------
// online softmax step for one 32-key block; MASKED=false skips causal compares
// (used for key blocks entirely below the diagonal for this wave's 16 rows)
// ---------------------------------------------------------------------------
#define PSTR 40
template<bool MASKED>
static __device__ __forceinline__ void softmax_step(
    v8f s[2], float* mrow, float* lrow, v8f acc[4],
    __bf16* sPw, int n0, int mloc0, int l16, int hl) {
    #pragma unroll
    for (int j = 0; j < 8; ++j) {
        float s0 = s[0][j] * 0.125f;
        float s1 = s[1][j] * 0.125f;
        if (MASKED) {
            int mg = mloc0 + j;
            if (n0 + l16 > mg)      s0 = -1e30f;
            if (n0 + 16 + l16 > mg) s1 = -1e30f;
        }
        float mb = fmaxf(s0, s1);
        #pragma unroll
        for (int o = 1; o < 16; o <<= 1) mb = fmaxf(mb, __shfl_xor(mb, o));
        float mn = fmaxf(mrow[j], mb);
        float alpha = __expf(mrow[j] - mn);
        mrow[j] = mn;
        lrow[j] *= alpha;
        acc[0][j] *= alpha; acc[1][j] *= alpha; acc[2][j] *= alpha; acc[3][j] *= alpha;
        float p0 = __expf(s0 - mn), p1 = __expf(s1 - mn);
        int r = (hl ? 8 : 0) + j;
        sPw[r * PSTR + l16]      = f2bf(p0);
        sPw[r * PSTR + 16 + l16] = f2bf(p1);
        float rs = p0 + p1;
        #pragma unroll
        for (int o = 1; o < 16; o <<= 1) rs += __shfl_xor(rs, o);
        lrow[j] += rs;
    }
}

// ---------------------------------------------------------------------------
// Fused flash attention. grid (H, T/128), 256 threads, wave owns 16 q rows.
// Q,K: (H,T,64) bf16.  Vt: (H,64,T) bf16.  O: (T,D) bf16 (head h -> cols h*64..)
// ---------------------------------------------------------------------------
__global__ __launch_bounds__(256)
void attn_fused(const __bf16* __restrict__ Q, const __bf16* __restrict__ Kb,
                const __bf16* __restrict__ Vt, __bf16* __restrict__ O) {
    __shared__ __bf16 sP[8][16 * PSTR];
    const int tid = threadIdx.x, lane = tid & 31, wave = tid >> 5;
    const int hl = lane >> 4, l16 = lane & 15;
    const int h = blockIdx.x;
    const int qBase = blockIdx.y * 128;
    const int rowBase = qBase + wave * 16;

    const __bf16* qrow = Q + ((size_t)h * T_SEQ + rowBase + l16) * HDIM;
    v16bf qa0 = frag16(qrow + hl * 8,      qrow + 16 + hl * 8);   // K 0..31
    v16bf qa1 = frag16(qrow + 32 + hl * 8, qrow + 48 + hl * 8);   // K 32..63

    v8f acc[4];
    #pragma unroll
    for (int d = 0; d < 4; ++d)
        #pragma unroll
        for (int j = 0; j < 8; ++j) acc[d][j] = 0.f;
    float mrow[8], lrow[8];
    #pragma unroll
    for (int j = 0; j < 8; ++j) { mrow[j] = -1e30f; lrow[j] = 0.f; }

    __bf16* sPw = sP[wave];
    const int mloc0 = rowBase + (hl ? 8 : 0);
    const int nb = (qBase + 128) >> 5;   // causal: keys < qBase+128
    for (int kb = 0; kb < nb; ++kb) {
        int n0 = kb * 32;
        v8f s[2];
        #pragma unroll
        for (int ni = 0; ni < 2; ++ni) {
            #pragma unroll
            for (int j = 0; j < 8; ++j) s[ni][j] = 0.f;
            const __bf16* krow = Kb + ((size_t)h * T_SEQ + n0 + ni * 16 + l16) * HDIM;
            v16bf kf0 = frag16(krow + hl * 8,      krow + 16 + hl * 8);
            v16bf kf1 = frag16(krow + 32 + hl * 8, krow + 48 + hl * 8);
            s[ni] = __builtin_amdgcn_wmma_f32_16x16x32_bf16(false, qa0, false, kf0, (short)0, s[ni], false, false);
            s[ni] = __builtin_amdgcn_wmma_f32_16x16x32_bf16(false, qa1, false, kf1, (short)0, s[ni], false, false);
        }
        if (n0 + 32 <= rowBase)
            softmax_step<false>(s, mrow, lrow, acc, sPw, n0, mloc0, l16, hl);
        else
            softmax_step<true>(s, mrow, lrow, acc, sPw, n0, mloc0, l16, hl);

        asm volatile("s_wait_dscnt 0" ::: "memory");   // wave-local LDS write -> read
        v16bf pa = frag16(&sPw[l16 * PSTR + hl * 8], &sPw[l16 * PSTR + 16 + hl * 8]);
        #pragma unroll
        for (int d = 0; d < 4; ++d) {
            const __bf16* vrow = Vt + ((size_t)h * HDIM + d * 16 + l16) * T_SEQ + n0;
            v16bf vf = frag16(vrow + hl * 8, vrow + 16 + hl * 8);
            acc[d] = __builtin_amdgcn_wmma_f32_16x16x32_bf16(false, pa, false, vf, (short)0, acc[d], false, false);
        }
    }
    #pragma unroll
    for (int d = 0; d < 4; ++d) {
        int c = h * HDIM + d * 16 + l16;
        #pragma unroll
        for (int j = 0; j < 8; ++j) {
            int m = rowBase + (hl ? 8 : 0) + j;
            O[(size_t)m * D_MODEL + c] = f2bf(acc[d][j] / lrow[j]);
        }
    }
}

// ---------------------------------------------------------------------------
// final: column mean of (x * inv_rms[t]) * w, then L2 normalize
// ---------------------------------------------------------------------------
__global__ void colmean_k(const float* __restrict__ X, const float* __restrict__ inv,
                          const float* __restrict__ W, float* __restrict__ Hm) {
    int d = blockIdx.x * 256 + threadIdx.x;
    if (d >= D_MODEL) return;
    float s = 0.f;
    for (int t = 0; t < T_SEQ; ++t) s += X[(size_t)t * D_MODEL + d] * inv[t];
    Hm[d] = (s * (1.f / (float)T_SEQ)) * W[d];
}

__global__ __launch_bounds__(256)
void l2_final(const float* __restrict__ Hm, float* __restrict__ Out) {
    __shared__ float red[8];
    int tid = threadIdx.x;
    float v0 = Hm[tid], v1 = Hm[tid + 256], v2 = Hm[tid + 512];
    float ss = v0 * v0 + v1 * v1 + v2 * v2;
    #pragma unroll
    for (int o = 16; o; o >>= 1) ss += __shfl_xor(ss, o);
    if ((tid & 31) == 0) red[tid >> 5] = ss;
    __syncthreads();
    float tot = 0.f;
    #pragma unroll
    for (int i = 0; i < 8; ++i) tot += red[i];
    float n = fmaxf(sqrtf(tot), 1e-12f);
    Out[tid]       = v0 / n;
    Out[tid + 256] = v1 / n;
    Out[tid + 512] = v2 / n;
}

// ---------------------------------------------------------------------------
extern "C" void kernel_launch(void* const* d_in, const int* in_sizes, int n_in,
                              void* d_out, int out_size, void* d_ws, size_t ws_size,
                              hipStream_t stream) {
    (void)in_sizes; (void)n_in; (void)out_size; (void)ws_size;
    const int*   ids  = (const int*)d_in[0];
    const float* emb  = (const float*)d_in[1];
    const float* pos  = (const float*)d_in[2];
    const float* qkvw = (const float*)d_in[3];
    const float* outw = (const float*)d_in[4];
    const float* w1   = (const float*)d_in[5];
    const float* w2   = (const float*)d_in[6];
    const float* n1   = (const float*)d_in[7];
    const float* n2   = (const float*)d_in[8];
    const float* fnw  = (const float*)d_in[9];
    float* out = (float*)d_out;

    char* ws = (char*)d_ws;
    size_t off = 0;
    auto carve = [&](size_t bytes) -> char* {
        char* p = ws + off;
        off += (bytes + 255) & ~(size_t)255;
        return p;
    };
    // per-layer bf16 weights (converted on the fly)
    __bf16* qkvw_b = (__bf16*)carve((size_t)3 * D_MODEL * D_MODEL * 2);
    __bf16* outw_b = (__bf16*)carve((size_t)D_MODEL * D_MODEL * 2);
    __bf16* w1_b   = (__bf16*)carve((size_t)FFDIM * D_MODEL * 2);
    __bf16* w2_b   = (__bf16*)carve((size_t)D_MODEL * FFDIM * 2);
    // activations
    float*  x    = (float*)carve((size_t)T_SEQ * D_MODEL * 4);
    __bf16* hbuf = (__bf16*)carve((size_t)T_SEQ * D_MODEL * 2);
    float*  qkv  = (float*)carve((size_t)T_SEQ * 3 * D_MODEL * 4);
    __bf16* qh   = (__bf16*)carve((size_t)NHEAD * T_SEQ * HDIM * 2);
    __bf16* kh   = (__bf16*)carve((size_t)NHEAD * T_SEQ * HDIM * 2);
    __bf16* vt   = (__bf16*)carve((size_t)NHEAD * HDIM * T_SEQ * 2);
    __bf16* ao   = (__bf16*)carve((size_t)T_SEQ * D_MODEL * 2);
    __bf16* ff   = (__bf16*)carve((size_t)T_SEQ * FFDIM * 2);
    float*  invb = (float*)carve((size_t)T_SEQ * 4);
    float*  hm   = (float*)carve((size_t)D_MODEL * 4);

    embed_k<<<(T_SEQ * D_MODEL) / 256, 256, 0, stream>>>(ids, emb, pos, x);

    const long nQKV = (long)3 * D_MODEL * D_MODEL;
    const long nOUT = (long)D_MODEL * D_MODEL;
    const long nW1  = (long)FFDIM * D_MODEL;

    for (int l = 0; l < NLAYER; ++l) {
        // convert this layer's weights to bf16
        cvt_bf16<<<(int)((nQKV / 8 + 255) / 256), 256, 0, stream>>>(qkvw + (size_t)l * nQKV, qkvw_b, nQKV / 8);
        cvt_bf16<<<(int)((nOUT / 8 + 255) / 256), 256, 0, stream>>>(outw + (size_t)l * nOUT, outw_b, nOUT / 8);
        cvt_bf16<<<(int)((nW1 / 8 + 255) / 256), 256, 0, stream>>>(w1 + (size_t)l * nW1, w1_b, nW1 / 8);
        cvt_bf16<<<(int)((nW1 / 8 + 255) / 256), 256, 0, stream>>>(w2 + (size_t)l * nW1, w2_b, nW1 / 8);

        // attention half
        rmsnorm_bf16<<<T_SEQ, 256, 0, stream>>>(x, n1 + (size_t)l * D_MODEL, hbuf);
        gemm_bf16<0><<<dim3((3 * D_MODEL) / 64, T_SEQ / 128), 256, 0, stream>>>(
            hbuf, qkvw_b, qkv, nullptr, nullptr, 3 * D_MODEL, D_MODEL);
        qkv_post<<<T_SEQ * NHEAD, 64, 0, stream>>>(qkv, qh, kh, vt);
        attn_fused<<<dim3(NHEAD, T_SEQ / 128), 256, 0, stream>>>(qh, kh, vt, ao);
        gemm_bf16<1><<<dim3(D_MODEL / 64, T_SEQ / 128), 256, 0, stream>>>(
            ao, outw_b, x, nullptr, x, D_MODEL, D_MODEL);

        // FFN half
        rmsnorm_bf16<<<T_SEQ, 256, 0, stream>>>(x, n2 + (size_t)l * D_MODEL, hbuf);
        gemm_bf16<2><<<dim3(FFDIM / 64, T_SEQ / 128), 256, 0, stream>>>(
            hbuf, w1_b, nullptr, ff, nullptr, FFDIM, D_MODEL);
        gemm_bf16<1><<<dim3(D_MODEL / 64, T_SEQ / 128), 256, 0, stream>>>(
            ff, w2_b, x, nullptr, x, D_MODEL, FFDIM);
    }

    invrms_k<<<T_SEQ, 256, 0, stream>>>(x, invb);
    colmean_k<<<3, 256, 0, stream>>>(x, invb, fnw, hm);
    l2_final<<<1, 256, 0, stream>>>(hm, out);
}
